// DialogueGCN_FG_163208757769
// MI455X (gfx1250) — matrix operand order repaired and checked by reading, not compiled, not run on hardware
//
#include <hip/hip_runtime.h>
#include <hip/hip_bf16.h>

// ---------------------------------------------------------------------------
// CDNA5 (gfx1250) DialogueGCN forward: all GEMMs via v_wmma_f32_16x16x32_bf16,
// bf16 tile staging via global_load_async_to_lds_b128 (ASYNCcnt DMA).
// ---------------------------------------------------------------------------

typedef __bf16 bf16;
typedef __attribute__((ext_vector_type(16))) __bf16 v16bf;
typedef __attribute__((ext_vector_type(8)))  __bf16 v8bf;
typedef __attribute__((ext_vector_type(8)))  float  v8f;

#define DEV __device__ __forceinline__

// A fragment: 16x32 bf16 tile, row-major with row stride ld (elements).
// ISA layout: lane half h=(lane>>4), row m=lane&15; element j -> k=16*(j>>3)+8h+(j&7).
// Per lane that is two contiguous 8-element (16B) runs: [m*ld+8h] and [m*ld+16+8h].
DEV v16bf frag_a(const bf16* p, int ld, int lane) {
  int h = lane >> 4, m = lane & 15;
  const bf16* r0 = p + m * ld + h * 8;
  v8bf lo = *(const v8bf*)r0;
  v8bf hi = *(const v8bf*)(r0 + 16);
  return __builtin_shufflevector(lo, hi, 0, 1, 2, 3, 4, 5, 6, 7,
                                 8, 9, 10, 11, 12, 13, 14, 15);
}

// B fragment from TRANSPOSED storage [N][K] row-major (stride ld elements).
// ISA layout: n=lane&15, element j -> k=16*(j>>3)+8h+(j&7) (same as frag_a).
DEV v16bf frag_bt(const bf16* p, int ld, int lane) {
  const bf16* r0 = p + (lane & 15) * ld + (lane >> 4) * 8;
  v8bf lo = *(const v8bf*)r0;
  v8bf hi = *(const v8bf*)(r0 + 16);
  return __builtin_shufflevector(lo, hi, 0, 1, 2, 3, 4, 5, 6, 7,
                                 8, 9, 10, 11, 12, 13, 14, 15);
}

DEV v8f wmma_bf16(v16bf a, v16bf b, v8f c) {
  return __builtin_amdgcn_wmma_f32_16x16x32_bf16(
      /*neg_a=*/false, a, /*neg_b=*/false, b,
      /*c_mod=*/(short)0, c, /*reuse_a=*/false, /*reuse_b=*/false);
}

// Async DMA: 16 bytes global -> LDS, tracked by ASYNCcnt.
DEV void async_b128(unsigned int lds_off, const void* gaddr) {
  asm volatile("global_load_async_to_lds_b128 %0, %1, off"
               :: "v"(lds_off), "v"(gaddr) : "memory");
}
DEV void wait_async0() {
  asm volatile("s_wait_asynccnt 0" ::: "memory");
}
DEV unsigned int lds_off(const void* p) {
  return (unsigned int)(unsigned long long)p;  // low 32 bits = LDS byte address
}

// ---------------------------------------------------------------------------
// Kernel 1: Bahdanau global attention -> gw[256]   (tiny; VALU)
// ---------------------------------------------------------------------------
__global__ void __launch_bounds__(256) k_gatt(
    const float* __restrict__ GF, const float* __restrict__ Wq,
    const float* __restrict__ Wk, const float* __restrict__ v_att,
    float* __restrict__ gw) {
  __shared__ float qs[16][128];
  __shared__ float ks[16][128];
  __shared__ float sc[16][16];
  int tid = threadIdx.x;
  for (int r = tid; r < 16 * 128; r += 256) {
    int i = r >> 7, a = r & 127;
    float sq = 0.f, sk = 0.f;
    for (int d = 0; d < 256; ++d) {
      float g = GF[i * 256 + d];
      sq += g * Wq[d * 128 + a];
      sk += g * Wk[d * 128 + a];
    }
    qs[i][a] = sq; ks[i][a] = sk;
  }
  __syncthreads();
  {
    int i = tid >> 4, j = tid & 15;
    float s = 0.f;
    for (int a = 0; a < 128; ++a) s += v_att[a] * tanhf(qs[i][a] + ks[j][a]);
    sc[i][j] = s;
  }
  __syncthreads();
  if (tid < 16) {
    float m = -1e30f;
    for (int j = 0; j < 16; ++j) m = fmaxf(m, sc[tid][j]);
    float e[16], sum = 0.f;
    for (int j = 0; j < 16; ++j) { e[j] = __expf(sc[tid][j] - m); sum += e[j]; }
    float inv = 1.f / sum;
    for (int j = 0; j < 16; ++j) gw[tid * 16 + j] = e[j] * inv;
  }
}

// ---------------------------------------------------------------------------
// Kernel 2: relation ids per edge
// ---------------------------------------------------------------------------
__global__ void k_rel(const int* __restrict__ spk, int* __restrict__ rel) {
  int e = threadIdx.x;
  int i = e >> 4, j = e & 15;
  int dir = (i < j) ? 0 : 1;
  rel[e] = (spk[i] * 16 + spk[j]) * 2 + dir;
}

// ---------------------------------------------------------------------------
// Kernel 3: fp32 -> bf16 conversion (grid-stride)
// ---------------------------------------------------------------------------
__global__ void k_cvt(const float* __restrict__ X, bf16* __restrict__ Y, int n) {
  int idx = blockIdx.x * blockDim.x + threadIdx.x;
  for (; idx < n; idx += gridDim.x * blockDim.x) Y[idx] = (bf16)X[idx];
}

// ---------------------------------------------------------------------------
// Kernel 4: P = X[2048x256] @ W[256x128] -> bf16  (WMMA)
// block = 256 thr = 8 waves in 2x4 tile grid; block tile 128x128
// ---------------------------------------------------------------------------
__global__ void __launch_bounds__(256, 1) k_proj(
    const float* __restrict__ X, const float* __restrict__ W,
    bf16* __restrict__ P) {
  __shared__ alignas(64) bf16 As[128][32];   // [m][k]
  __shared__ alignas(64) bf16 Bt[128][32];   // [n][k] (transposed for frag_bt)
  int tid = threadIdx.x, lane = tid & 31, w = tid >> 5;
  int wm = w & 3, wn = w >> 2;
  int row0 = blockIdx.x * 128;
  v8f c[2][4];
#pragma unroll
  for (int a = 0; a < 2; ++a)
#pragma unroll
    for (int b = 0; b < 4; ++b) c[a][b] = {};

  for (int kt = 0; kt < 8; ++kt) {
    for (int r = tid; r < 128 * 32; r += 256) {
      int m = r >> 5, k = r & 31;                 // consecutive tid -> consecutive k
      As[m][k] = (bf16)X[(size_t)(row0 + m) * 256 + kt * 32 + k];
    }
    for (int r = tid; r < 32 * 128; r += 256) {
      int k = r >> 7, n = r & 127;                // consecutive tid -> consecutive n (coalesced)
      Bt[n][k] = (bf16)W[(kt * 32 + k) * 128 + n];
    }
    __syncthreads();
    v16bf a0 = frag_a(&As[wm * 32][0], 32, lane);
    v16bf a1 = frag_a(&As[wm * 32 + 16][0], 32, lane);
#pragma unroll
    for (int ni = 0; ni < 4; ++ni) {
      v16bf bfg = frag_bt(&Bt[wn * 64 + ni * 16][0], 32, lane);
      c[0][ni] = wmma_bf16(a0, bfg, c[0][ni]);
      c[1][ni] = wmma_bf16(a1, bfg, c[1][ni]);
    }
    __syncthreads();
  }
  int n = lane & 15, mh = (lane >> 4) << 3;
#pragma unroll
  for (int mi = 0; mi < 2; ++mi)
#pragma unroll
    for (int ni = 0; ni < 4; ++ni)
#pragma unroll
      for (int r = 0; r < 8; ++r) {
        int m = row0 + wm * 32 + mi * 16 + mh + r;
        int col = wn * 64 + ni * 16 + n;
        P[(size_t)m * 128 + col] = (bf16)c[mi][ni][r];
      }
}

// ---------------------------------------------------------------------------
// Kernel 5: per-edge attention S = p1[i] @ p2[j]^T, masked softmax,
//           store lw^T[e][s][t] and ew^T[e][s][t] (pre-transposed for RGCN A).
//           p1/p2 tiles staged with async-to-LDS DMA (pure bf16 copies).
// ---------------------------------------------------------------------------
__global__ void __launch_bounds__(256, 1) k_pair(
    const bf16* __restrict__ P1, const bf16* __restrict__ P2,
    const float* __restrict__ gw, const int* __restrict__ length,
    bf16* __restrict__ lwT, bf16* __restrict__ ewT) {
  __shared__ alignas(64) bf16 Ab[128][32];   // p1[i] chunk [t][a]
  __shared__ alignas(64) bf16 Bb[128][32];   // p2[j] chunk [s][a] == transposed B
  __shared__ alignas(64) bf16 Sb[128][128];
  int tid = threadIdx.x, lane = tid & 31, w = tid >> 5;
  int e = blockIdx.x, i = e >> 4, j = e & 15;
  v8f c[8];
#pragma unroll
  for (int f = 0; f < 8; ++f) c[f] = {};

  unsigned int abase = lds_off(&Ab[0][0]);
  unsigned int bbase = lds_off(&Bb[0][0]);
  for (int kt = 0; kt < 4; ++kt) {
    // 512 x 16B chunks per buffer; thread -> 2 chunks each buffer
    for (int cidx = tid; cidx < 512; cidx += 256) {
      int t = cidx >> 2, q = cidx & 3;
      const bf16* ga = P1 + ((size_t)i * 128 + t) * 128 + kt * 32 + q * 8;
      const bf16* gb = P2 + ((size_t)j * 128 + t) * 128 + kt * 32 + q * 8;
      async_b128(abase + t * 64 + q * 16, ga);
      async_b128(bbase + t * 64 + q * 16, gb);
    }
    wait_async0();
    __syncthreads();
    v16bf af = frag_a(&Ab[w * 16][0], 32, lane);
#pragma unroll
    for (int f = 0; f < 8; ++f) {
      v16bf bfg = frag_bt(&Bb[f * 16][0], 32, lane);  // contract over a
      c[f] = wmma_bf16(af, bfg, c[f]);
    }
    __syncthreads();
  }
  int n = lane & 15, mh = (lane >> 4) << 3;
#pragma unroll
  for (int f = 0; f < 8; ++f)
#pragma unroll
    for (int r = 0; r < 8; ++r)
      Sb[w * 16 + mh + r][f * 16 + n] = (bf16)c[f][r];
  __syncthreads();

  if (tid < 128) {
    int t = tid;
    int leni = length[i], lenj = length[j];
    float ge = gw[e];
    float mx = -1e30f;
    for (int s = 0; s < lenj; ++s) mx = fmaxf(mx, (float)Sb[t][s]);
    float sum = 0.f;
    for (int s = 0; s < lenj; ++s) sum += __expf((float)Sb[t][s] - mx);
    float inv = (t < leni) ? (1.f / sum) : 0.f;
    for (int s = 0; s < 128; ++s) {
      float wv = (s < lenj) ? __expf((float)Sb[t][s] - mx) * inv : 0.f;
      size_t o = ((size_t)e * 128 + s) * 128 + t;  // transposed store [e][s][t]
      lwT[o] = (bf16)wv;
      ewT[o] = (bf16)(ge * wv);
    }
  }
}

// ---------------------------------------------------------------------------
// Kernel 6: RGCN layer. One block per dst node j; 16 incoming edges.
//   per edge: panel-fused  (WT[e] @ X[src]) @ Wrel[rel[e]]  via WMMA,
//   accumulated in registers (no atomics); then + X[j]@Wroot + b.
// ---------------------------------------------------------------------------
__global__ void __launch_bounds__(256, 1) k_rgcn(
    const bf16* __restrict__ WT,     // [256][128][128]  (e, s, t)
    const bf16* __restrict__ Xbf,    // [16][128][256]
    const float* __restrict__ Wrel,  // [R][256][256]
    const int* __restrict__ rel,     // [256] or nullptr (-> relation 0)
    const float* __restrict__ Wroot, // [256][256]
    const float* __restrict__ bias,  // [256]
    float* __restrict__ Yout,        // [16][128][256] or nullptr
    bf16* __restrict__ Ybf) {        // [16][128][256] or nullptr
  __shared__ alignas(64) bf16 WTst[128][32];   // A tile [s][t-chunk]
  __shared__ alignas(64) bf16 Xt[32][32];      // B tile transposed [d][k]
  __shared__ alignas(64) bf16 Ach[128][32];    // GEMM1 panel [s][d-chunk]
  __shared__ alignas(64) bf16 West[256][32];   // GEMM2 B transposed [n][k]
  int tid = threadIdx.x, lane = tid & 31, w = tid >> 5;
  int j = blockIdx.x;
  int n = lane & 15, mh = (lane >> 4) << 3;
  unsigned int wtbase = lds_off(&WTst[0][0]);
  v8f acc[16];
#pragma unroll
  for (int f = 0; f < 16; ++f) acc[f] = {};

  for (int i = 0; i < 16; ++i) {
    int e = i * 16 + j;
    int rl = rel ? rel[e] : 0;
    const float* We = Wrel + (size_t)rl * 65536;
    const bf16* WTe = WT + (size_t)e * 16384;
    const bf16* Xi  = Xbf + (size_t)i * 32768;
    for (int dc = 0; dc < 8; ++dc) {       // 32-wide column panel of GEMM1
      v8f g0 = {}, g1 = {};
      for (int kt = 0; kt < 4; ++kt) {     // K = 128 (src tokens)
        // edge-weight tile: straight bf16 copy -> async DMA to LDS
        for (int cidx = tid; cidx < 512; cidx += 256) {
          int s = cidx >> 2, q = cidx & 3;
          async_b128(wtbase + s * 64 + q * 16,
                     WTe + s * 128 + kt * 32 + q * 8);
        }
        // X tile, transposed into [d][k] (coalesced global reads)
        for (int r = tid; r < 1024; r += 256) {
          int k = r >> 5, d = r & 31;
          Xt[d][k] = Xi[(kt * 32 + k) * 256 + dc * 32 + d];
        }
        wait_async0();
        __syncthreads();
        v16bf af = frag_a(&WTst[w * 16][0], 32, lane);
        v16bf b0 = frag_bt(&Xt[0][0], 32, lane);
        v16bf b1 = frag_bt(&Xt[16][0], 32, lane);
        g0 = wmma_bf16(af, b0, g0);
        g1 = wmma_bf16(af, b1, g1);
        __syncthreads();
      }
      // panel -> LDS bf16 (each wave writes only its own 16 rows)
#pragma unroll
      for (int r = 0; r < 8; ++r) {
        Ach[w * 16 + mh + r][n]      = (bf16)g0[r];
        Ach[w * 16 + mh + r][16 + n] = (bf16)g1[r];
      }
      // stage We rows [dc*32, dc*32+32) x 256, fp32 -> bf16, transposed [n][k]
      if (dc < 7) __builtin_prefetch(We + (size_t)(dc * 32 + 32) * 256, 0, 1);
      for (int r = tid; r < 8192; r += 256) {
        int k = r >> 8, nn = r & 255;               // consecutive tid -> consecutive nn
        West[nn][k] = (bf16)We[(size_t)(dc * 32 + k) * 256 + nn];
      }
      __syncthreads();
      // GEMM2 partial: acc += Ach(128x32) @ We(32x256)
      v16bf af2 = frag_a(&Ach[w * 16][0], 32, lane);
#pragma unroll
      for (int f = 0; f < 16; ++f) {
        v16bf bfg = frag_bt(&West[f * 16][0], 32, lane);
        acc[f] = wmma_bf16(af2, bfg, acc[f]);
      }
      __syncthreads();
    }
  }

  // root term: acc += X[j] @ Wroot
  for (int kt = 0; kt < 8; ++kt) {
    for (int r = tid; r < 8192; r += 256) {
      int k = r >> 8, nn = r & 255;
      West[nn][k] = (bf16)Wroot[(size_t)(kt * 32 + k) * 256 + nn];
    }
    __syncthreads();
    v16bf af = frag_a(Xbf + (size_t)j * 32768 + (size_t)(w * 16) * 256 + kt * 32,
                      256, lane);
#pragma unroll
    for (int f = 0; f < 16; ++f) {
      v16bf bfg = frag_bt(&West[f * 16][0], 32, lane);
      acc[f] = wmma_bf16(af, bfg, acc[f]);
    }
    __syncthreads();
  }

  // bias + store
#pragma unroll
  for (int f = 0; f < 16; ++f) {
    int col = f * 16 + n;
    float bv = bias[col];
#pragma unroll
    for (int r = 0; r < 8; ++r) {
      int m = w * 16 + mh + r;
      float v = acc[f][r] + bv;
      size_t o = (size_t)j * 32768 + (size_t)m * 256 + col;
      if (Yout) Yout[o] = v;
      if (Ybf)  Ybf[o]  = (bf16)v;
    }
  }
}

// ---------------------------------------------------------------------------
// Host launcher
// ---------------------------------------------------------------------------
extern "C" void kernel_launch(void* const* d_in, const int* in_sizes, int n_in,
                              void* d_out, int out_size, void* d_ws, size_t ws_size,
                              hipStream_t stream) {
  const float* GF     = (const float*)d_in[0];
  const float* LF     = (const float*)d_in[1];
  const int*   speaker= (const int*)d_in[2];
  const int*   length = (const int*)d_in[3];
  const float* Wq     = (const float*)d_in[4];
  const float* Wk     = (const float*)d_in[5];
  const float* v_att  = (const float*)d_in[6];
  const float* Wb1    = (const float*)d_in[7];
  const float* Wb2    = (const float*)d_in[8];
  const float* Wrel1  = (const float*)d_in[9];
  const float* Wroot1 = (const float*)d_in[10];
  const float* b1     = (const float*)d_in[11];
  const float* Wrel2  = (const float*)d_in[12];
  const float* Wroot2 = (const float*)d_in[13];
  const float* b2     = (const float*)d_in[14];

  char* ws = (char*)d_ws;
  size_t off = 0;
  auto alloc = [&](size_t bytes) -> void* {
    void* p = ws + off;
    off = (off + bytes + 255) & ~(size_t)255;
    return p;
  };
  float* gw   = (float*)alloc(256 * sizeof(float));
  int*   rel  = (int*)alloc(256 * sizeof(int));
  bf16*  xbf  = (bf16*)alloc((size_t)16 * 128 * 256 * sizeof(bf16));
  bf16*  p1   = (bf16*)alloc((size_t)16 * 128 * 128 * sizeof(bf16));
  bf16*  p2   = (bf16*)alloc((size_t)16 * 128 * 128 * sizeof(bf16));
  bf16*  lwT  = (bf16*)alloc((size_t)256 * 128 * 128 * sizeof(bf16));
  bf16*  ewT  = (bf16*)alloc((size_t)256 * 128 * 128 * sizeof(bf16));
  bf16*  x1bf = (bf16*)alloc((size_t)16 * 128 * 256 * sizeof(bf16));

  k_gatt<<<1, 256, 0, stream>>>(GF, Wq, Wk, v_att, gw);
  k_rel<<<1, 256, 0, stream>>>(speaker, rel);
  k_cvt<<<512, 256, 0, stream>>>(LF, xbf, 16 * 128 * 256);
  k_proj<<<16, 256, 0, stream>>>(LF, Wb1, p1);
  k_proj<<<16, 256, 0, stream>>>(LF, Wb2, p2);
  k_pair<<<256, 256, 0, stream>>>(p1, p2, gw, length, lwT, ewT);
  // layer 1: edge weights ew, per-edge relation, input = local features
  k_rgcn<<<16, 256, 0, stream>>>(ewT, xbf, Wrel1, rel, Wroot1, b1,
                                 nullptr, x1bf);
  // layer 2: edge weights lw, single relation 0, input = x1
  k_rgcn<<<16, 256, 0, stream>>>(lwT, x1bf, Wrel2, nullptr, Wroot2, b2,
                                 (float*)d_out, nullptr);
}
